// VMF_25804163514999
// MI455X (gfx1250) — matrix operand order repaired and checked by reading, not compiled, not run on hardware
//
#include <hip/hip_runtime.h>

typedef float  v8f  __attribute__((ext_vector_type(8)));
typedef __bf16 v8bf __attribute__((ext_vector_type(8)));
typedef __bf16 v16bf __attribute__((ext_vector_type(16)));

#define H_DIM 1024
#define D_DIM 512
#define KC 32           // K per stage = one 16x16x32 bf16 wmma k-step
#define LDP 20          // LDS row stride in dwords (16 data + 4 pad) -> conflict-free b128
#define MBLK 64         // rows per workgroup
#define NSTAGES (H_DIM / KC)

// ---- fp32 <-> bf16 bit helpers (RNE), no scalar __bf16 arithmetic needed ----
__device__ __forceinline__ unsigned f2bf(float x) {
    unsigned u = __builtin_bit_cast(unsigned, x);
    return (u + 0x7fffu + ((u >> 16) & 1u)) >> 16;
}
__device__ __forceinline__ float bf2f(unsigned h) {
    return __builtin_bit_cast(float, h << 16);
}
__device__ __forceinline__ unsigned packbf(unsigned lo16, unsigned hi16) {
    return (lo16 & 0xffffu) | (hi16 << 16);
}
__device__ __forceinline__ float bflo(float x) {
    return x - bf2f(f2bf(x));
}

__device__ __forceinline__ unsigned mixhash(unsigned x) {
    x ^= x >> 16; x *= 0x7feb352du;
    x ^= x >> 15; x *= 0x846ca68bu;
    x ^= x >> 16;
    return x;
}

__device__ __forceinline__ float gaussf(unsigned key) {
    unsigned h1 = mixhash(key * 2u + 0x9E3779B9u);
    unsigned h2 = mixhash(key * 2u + 0x85EBCA6Bu);
    float u1 = (float)(h1 >> 8) * (1.0f / 16777216.0f) + (0.5f / 16777216.0f);
    float u2 = (float)(h2 >> 8) * (1.0f / 16777216.0f);
    return sqrtf(-2.0f * __logf(u1)) * __cosf(6.28318530718f * u2);
}

// Wood's rejection sampler for vMF radial component (kappa=80, D=512).
__device__ float sample_w_row(unsigned row) {
    const float kappa = 80.0f, dim = 511.0f;
    const float bb = dim / (sqrtf(4.0f * kappa * kappa + dim * dim) + 2.0f * kappa);
    const float xx = (1.0f - bb) / (1.0f + bb);
    const float cc = kappa * xx + dim * __logf(1.0f - xx * xx);
    const float sigma = 0.0220970869f;  // 1/sqrt(2048)
    float w = 2.0f * bb / (1.0f + bb);
    bool done = false;
    #pragma unroll 4
    for (int r = 0; r < 64; ++r) {
        unsigned base = 0xC0FFEE00u + row * 131u + (unsigned)r * 0x01000193u;
        float g = gaussf(base);
        float z = fminf(fmaxf(0.5f + sigma * g, 1e-6f), 1.0f - 1e-6f);
        unsigned hu = mixhash(base + 0x6b43a9b5u);
        float u = (float)(hu >> 8) * (1.0f / 16777216.0f) + (0.5f / 16777216.0f);
        float wc = (1.0f - (1.0f + bb) * z) / (1.0f - (1.0f - bb) * z);
        bool acc = (kappa * wc + dim * __logf(1.0f - xx * wc) - cc) >= __logf(u);
        if (acc && !done) { w = wc; done = true; }
    }
    return w;
}

__device__ __forceinline__ float softplusf(float x) {
    return (x > 20.0f) ? x : log1pf(__expf(x));
}

__device__ __forceinline__ v16bf ldfrag(const unsigned* p0, const unsigned* p1) {
    v8bf a = *(const v8bf*)p0;
    v8bf b = *(const v8bf*)p1;
    return __builtin_shufflevector(a, b, 0,1,2,3,4,5,6,7,8,9,10,11,12,13,14,15);
}

// Async copy of 16 bytes from global memory into LDS (per lane), ASYNCcnt-tracked.
__device__ __forceinline__ void async_g2l_b128(unsigned lds_byte_off, const void* gptr) {
    unsigned long long ga = (unsigned long long)gptr;
    asm volatile("global_load_async_to_lds_b128 %0, %1, off"
                 :: "v"(lds_byte_off), "v"(ga) : "memory");
}
__device__ __forceinline__ void wait_asynccnt0() {
    asm volatile("s_wait_asynccnt 0x0" ::: "memory");
}

__global__ __launch_bounds__(512, 1)
void vmf_fused_kernel(const float* __restrict__ A,     // [B, 1024]
                      const float* __restrict__ Wg,    // [1024, 512]
                      const float* __restrict__ bias,  // [512]
                      const float* __restrict__ kld,   // [1]
                      float* __restrict__ out_mu,      // [B, 512]
                      float* __restrict__ out_rn,      // [B]
                      float* __restrict__ out_kld,     // [1]
                      float* __restrict__ out_sm,      // [B, 512]
                      int B) {
    __shared__ unsigned WtH[D_DIM * LDP];   // W^T hi bf16 pairs [512][16 dw]
    __shared__ unsigned WtL[D_DIM * LDP];   // W^T lo bf16 pairs
    __shared__ unsigned AshH[MBLK * LDP];   // A hi bf16 pairs, row-major [64][16 dw]
    __shared__ unsigned AshL[MBLK * LDP];   // A lo bf16 pairs
    __shared__ float AshRaw[2 * MBLK * KC]; // double-buffered raw fp32 A chunk (async dest)
    __shared__ float bsh[D_DIM];
    __shared__ float redA[MBLK], redB[MBLK], redC[MBLK];
    __shared__ float norms[MBLK], wsamp[MBLK];

    const int tid  = threadIdx.x;
    const int wave = tid >> 5;
    const int lane = tid & 31;
    const int lm   = lane & 15;
    const int lh   = lane >> 4;
    const int mtp  = wave & 1;            // row-tile pair: tiles {0,1} or {2,3}
    const int ntg  = wave >> 1;           // col-tile group: tiles ntg*4 .. ntg*4+3
    const int row0 = blockIdx.x * MBLK;

    if (tid < D_DIM) bsh[tid] = bias[tid];
    if (tid < MBLK) { redA[tid] = 0.0f; redB[tid] = 0.0f; redC[tid] = 0.0f; }

    v8f acc[2][4];
    #pragma unroll
    for (int rt = 0; rt < 2; ++rt)
        #pragma unroll
        for (int ct = 0; ct < 4; ++ct) acc[rt][ct] = (v8f){0,0,0,0,0,0,0,0};

    // A staging geometry: one float4 per thread per stage
    const int arow = tid >> 3;            // 0..63
    const int ak0  = (tid & 7) * 4;       // k 0,4,..,28
    const unsigned rawBase = (unsigned)(unsigned long long)(&AshRaw[0]); // LDS byte offset

    // prologue: async-stage raw A for stage 0 into buffer 0
    async_g2l_b128(rawBase + (unsigned)tid * 16u,
                   A + (size_t)(row0 + arow) * H_DIM + 0 * KC + ak0);

    for (int s = 0; s < NSTAGES; ++s) {
        const int buf = s & 1;
        wait_asynccnt0();       // this wave's async writes for stage s are in LDS
        __syncthreads();        // all waves' writes visible; prior compute done reading

        // ===== staging phase: convert A chunk, stage W hi+lo, kick next A async =====
        {
            const float4 a4 = *(const float4*)&AshRaw[buf * (MBLK * KC) + tid * 4];
            unsigned h0 = f2bf(a4.x), h1 = f2bf(a4.y), h2 = f2bf(a4.z), h3 = f2bf(a4.w);
            AshH[arow * LDP + (ak0 >> 1) + 0] = packbf(h0, h1);
            AshH[arow * LDP + (ak0 >> 1) + 1] = packbf(h2, h3);
            AshL[arow * LDP + (ak0 >> 1) + 0] = packbf(f2bf(a4.x - bf2f(h0)), f2bf(a4.y - bf2f(h1)));
            AshL[arow * LDP + (ak0 >> 1) + 1] = packbf(f2bf(a4.z - bf2f(h2)), f2bf(a4.w - bf2f(h3)));
        }
        if (s + 1 < NSTAGES) {  // pipeline: async-fill the other buffer for stage s+1
            async_g2l_b128(rawBase + (unsigned)(buf ^ 1) * (MBLK * KC * 4) + (unsigned)tid * 16u,
                           A + (size_t)(row0 + arow) * H_DIM + (s + 1) * KC + ak0);
        }
        #pragma unroll
        for (int u2 = 0; u2 < 2; ++u2) {
            const int u  = tid + u2 * 512;     // 0..1023
            const int q  = u >> 7;             // k-quad 0..7
            const int n0 = (u & 127) * 4;      // col group
            float wk[4][4];
            #pragma unroll
            for (int j = 0; j < 4; ++j) {
                const float4 w4 = *(const float4*)(Wg + (size_t)(s * KC + q * 4 + j) * D_DIM + n0);
                wk[j][0] = w4.x; wk[j][1] = w4.y; wk[j][2] = w4.z; wk[j][3] = w4.w;
            }
            if (s + 1 < NSTAGES)
                __builtin_prefetch(Wg + (size_t)((s + 1) * KC + q * 4) * D_DIM + n0, 0, 3);
            #pragma unroll
            for (int c = 0; c < 4; ++c) {
                WtH[(n0 + c) * LDP + q * 2 + 0] = packbf(f2bf(wk[0][c]), f2bf(wk[1][c]));
                WtH[(n0 + c) * LDP + q * 2 + 1] = packbf(f2bf(wk[2][c]), f2bf(wk[3][c]));
                WtL[(n0 + c) * LDP + q * 2 + 0] = packbf(f2bf(bflo(wk[0][c])), f2bf(bflo(wk[1][c])));
                WtL[(n0 + c) * LDP + q * 2 + 1] = packbf(f2bf(bflo(wk[2][c])), f2bf(bflo(wk[3][c])));
            }
        }
        __syncthreads();

        // ===== compute phase: Ah*Wh + Al*Wh + Ah*Wl, 2 row-tiles x 4 col-tiles =====
        v16bf ah[2], al[2];
        #pragma unroll
        for (int rt = 0; rt < 2; ++rt) {
            const int mrow = (mtp * 2 + rt) * 16 + lm;
            ah[rt] = ldfrag(&AshH[mrow * LDP + lh * 4], &AshH[mrow * LDP + 8 + lh * 4]);
            al[rt] = ldfrag(&AshL[mrow * LDP + lh * 4], &AshL[mrow * LDP + 8 + lh * 4]);
        }
        #pragma unroll
        for (int ct = 0; ct < 4; ++ct) {
            const int n = (ntg * 4 + ct) * 16 + lm;
            v16bf bh = ldfrag(&WtH[n * LDP + lh * 4], &WtH[n * LDP + 8 + lh * 4]);
            v16bf bl = ldfrag(&WtL[n * LDP + lh * 4], &WtL[n * LDP + 8 + lh * 4]);
            #pragma unroll
            for (int rt = 0; rt < 2; ++rt) {
                acc[rt][ct] = __builtin_amdgcn_wmma_f32_16x16x32_bf16(
                    false, ah[rt], false, bh, (short)0, acc[rt][ct], false, false);
                acc[rt][ct] = __builtin_amdgcn_wmma_f32_16x16x32_bf16(
                    false, al[rt], false, bh, (short)0, acc[rt][ct], false, false);
                acc[rt][ct] = __builtin_amdgcn_wmma_f32_16x16x32_bf16(
                    false, ah[rt], false, bl, (short)0, acc[rt][ct], false, false);
            }
        }
        __syncthreads();
    }

    // ---- epilogue: bias + softplus, row sum-of-squares ----
    float sumsq[2][8];
    #pragma unroll
    for (int rt = 0; rt < 2; ++rt)
        #pragma unroll
        for (int i = 0; i < 8; ++i) sumsq[rt][i] = 0.0f;
    #pragma unroll
    for (int ct = 0; ct < 4; ++ct) {
        const int n = (ntg * 4 + ct) * 16 + lm;
        const float bv = bsh[n];
        #pragma unroll
        for (int rt = 0; rt < 2; ++rt)
            #pragma unroll
            for (int i = 0; i < 8; ++i) {
                float sp = softplusf(acc[rt][ct][i] + bv);
                acc[rt][ct][i] = sp;            // acc now holds mu_raw
                sumsq[rt][i] += sp * sp;
            }
    }
    #pragma unroll
    for (int rt = 0; rt < 2; ++rt)
        #pragma unroll
        for (int i = 0; i < 8; ++i) {
            float v = sumsq[rt][i];
            v += __shfl_xor(v, 1, 32);
            v += __shfl_xor(v, 2, 32);
            v += __shfl_xor(v, 4, 32);
            v += __shfl_xor(v, 8, 32);
            if (lm == 0) atomicAdd(&redA[(mtp * 2 + rt) * 16 + i + lh * 8], v);
        }
    __syncthreads();

    if (tid < MBLK) {
        float nrm = sqrtf(redA[tid]);
        norms[tid] = nrm;
        out_rn[row0 + tid] = (nrm - 1.0f) * (nrm - 1.0f);
        wsamp[tid] = sample_w_row((unsigned)(row0 + tid));
    }
    if (blockIdx.x == 0 && tid == 0) out_kld[0] = kld[0];
    __syncthreads();

    // ---- normalize mu, write it, accumulate v.v and mu.v (v recomputed from hash) ----
    float pv2[2][8], pmv[2][8];
    #pragma unroll
    for (int rt = 0; rt < 2; ++rt)
        #pragma unroll
        for (int i = 0; i < 8; ++i) { pv2[rt][i] = 0.0f; pmv[rt][i] = 0.0f; }
    #pragma unroll
    for (int ct = 0; ct < 4; ++ct) {
        const int n = (ntg * 4 + ct) * 16 + lm;
        #pragma unroll
        for (int rt = 0; rt < 2; ++rt)
            #pragma unroll
            for (int i = 0; i < 8; ++i) {
                const int m  = (mtp * 2 + rt) * 16 + i + lh * 8;
                const float mu = acc[rt][ct][i] / norms[m];
                acc[rt][ct][i] = mu;            // acc now holds mu
                out_mu[(size_t)(row0 + m) * D_DIM + n] = mu;
                const float v = gaussf((unsigned)(row0 + m) * (unsigned)D_DIM + (unsigned)n);
                pv2[rt][i] += v * v;
                pmv[rt][i] += mu * v;
            }
    }
    #pragma unroll
    for (int rt = 0; rt < 2; ++rt)
        #pragma unroll
        for (int i = 0; i < 8; ++i) {
            float a = pv2[rt][i], b2 = pmv[rt][i];
            a  += __shfl_xor(a, 1, 32);  b2 += __shfl_xor(b2, 1, 32);
            a  += __shfl_xor(a, 2, 32);  b2 += __shfl_xor(b2, 2, 32);
            a  += __shfl_xor(a, 4, 32);  b2 += __shfl_xor(b2, 4, 32);
            a  += __shfl_xor(a, 8, 32);  b2 += __shfl_xor(b2, 8, 32);
            if (lm == 0) {
                atomicAdd(&redB[(mtp * 2 + rt) * 16 + i + lh * 8], a);
                atomicAdd(&redC[(mtp * 2 + rt) * 16 + i + lh * 8], b2);
            }
        }
    __syncthreads();

    // ---- sampled = alpha*v + beta*mu  (|ortho|^2 = |v|^2 - proj^2 since |mu|=1) ----
    float alpha[2][8], beta[2][8];
    #pragma unroll
    for (int rt = 0; rt < 2; ++rt)
        #pragma unroll
        for (int i = 0; i < 8; ++i) {
            const int m = (mtp * 2 + rt) * 16 + i + lh * 8;
            const float proj  = redC[m];
            const float vn2   = redB[m];
            const float w     = wsamp[m];
            const float onorm = sqrtf(fmaxf(vn2 - proj * proj, 1e-20f));
            alpha[rt][i] = sqrtf(fmaxf(1.0f - w * w, 0.0f)) / onorm;
            beta[rt][i]  = w - proj * alpha[rt][i];
        }
    #pragma unroll
    for (int ct = 0; ct < 4; ++ct) {
        const int n = (ntg * 4 + ct) * 16 + lm;
        #pragma unroll
        for (int rt = 0; rt < 2; ++rt)
            #pragma unroll
            for (int i = 0; i < 8; ++i) {
                const int m = (mtp * 2 + rt) * 16 + i + lh * 8;
                const float v = gaussf((unsigned)(row0 + m) * (unsigned)D_DIM + (unsigned)n);
                out_sm[(size_t)(row0 + m) * D_DIM + n] = alpha[rt][i] * v + beta[rt][i] * acc[rt][ct][i];
            }
    }
}

extern "C" void kernel_launch(void* const* d_in, const int* in_sizes, int n_in,
                              void* d_out, int out_size, void* d_ws, size_t ws_size,
                              hipStream_t stream) {
    const float* A    = (const float*)d_in[0];   // [B,1024]
    const float* Wg   = (const float*)d_in[1];   // [1024,512]
    const float* bias = (const float*)d_in[2];   // [512]
    const float* kld  = (const float*)d_in[3];   // [1]
    const int B = in_sizes[0] / H_DIM;

    float* out_mu  = (float*)d_out;
    float* out_rn  = out_mu + (size_t)B * D_DIM;
    float* out_kld = out_rn + B;
    float* out_sm  = out_kld + 1;

    dim3 grid(B / MBLK), block(512);
    vmf_fused_kernel<<<grid, block, 0, stream>>>(A, Wg, bias, kld,
                                                 out_mu, out_rn, out_kld, out_sm, B);
}
